// RecognitionModelV3Export_10857677324773
// MI455X (gfx1250) — matrix operand order repaired and checked by reading, not compiled, not run on hardware
//
#include <hip/hip_runtime.h>

#define kD 512
#define kB 16
#define kNH 8
#define kH (kB * kNH)   // 128 heads (b*NH+nh)
#define kFF 2048
#define kLE 6
#define kLD 6
#define kV 97
#define kSM 512
#define kSQ 128
#define kDK 64
#define kEPS 1e-5f

typedef __attribute__((ext_vector_type(16))) __bf16 v16bf;
typedef __attribute__((ext_vector_type(8))) float v8f;

__device__ __forceinline__ v8f wmma_bf16(v16bf a, v16bf b, v8f c) {
  return __builtin_amdgcn_wmma_f32_16x16x32_bf16(false, a, false, b, (short)0, c,
                                                 false, false);
}

__device__ __forceinline__ float redmax16(float v) {
  v = fmaxf(v, __shfl_xor(v, 1, 32));
  v = fmaxf(v, __shfl_xor(v, 2, 32));
  v = fmaxf(v, __shfl_xor(v, 4, 32));
  v = fmaxf(v, __shfl_xor(v, 8, 32));
  return v;
}
__device__ __forceinline__ float redsum16(float v) {
  v += __shfl_xor(v, 1, 32);
  v += __shfl_xor(v, 2, 32);
  v += __shfl_xor(v, 4, 32);
  v += __shfl_xor(v, 8, 32);
  return v;
}

// ---------------- f32 -> bf16 conversion ----------------
__global__ void cvt_kernel(const float* __restrict__ in, __bf16* __restrict__ out, int n) {
  int i = blockIdx.x * 256 + threadIdx.x;
  if (i < n) out[i] = (__bf16)in[i];
}

// Build a 16x32 bf16 A-fragment from a row-major f32 row pointer (row = lane&15).
// Layout: lane<16 -> K {kb..kb+15}; lane>=16 -> K {kb+16..kb+31} (matches contiguous B).
__device__ __forceinline__ v16bf afrag_from_f32(const float* p0) {
  const float* p1 = p0 + 8;
  float4 f0 = *(const float4*)(p0);
  float4 f1 = *(const float4*)(p0 + 4);
  float4 f2 = *(const float4*)(p1);
  float4 f3 = *(const float4*)(p1 + 4);
  v16bf a;
  a[0] = (__bf16)f0.x; a[1] = (__bf16)f0.y; a[2] = (__bf16)f0.z; a[3] = (__bf16)f0.w;
  a[4] = (__bf16)f1.x; a[5] = (__bf16)f1.y; a[6] = (__bf16)f1.z; a[7] = (__bf16)f1.w;
  a[8] = (__bf16)f2.x; a[9] = (__bf16)f2.y; a[10] = (__bf16)f2.z; a[11] = (__bf16)f2.w;
  a[12] = (__bf16)f3.x; a[13] = (__bf16)f3.y; a[14] = (__bf16)f3.z; a[15] = (__bf16)f3.w;
  return a;
}

// ---------------- generic WMMA GEMM: C[M,N] = A[M,K] @ W[N,K]^T + bias ----------------
// Block: 256 threads = 8 waves; block tile 128(M) x 128(N); wave tile 32(M) x 64(N).
// Two A fragments share each group of four B fragments (8 wmma per B-load group).
__global__ void gemm_kernel(const float* __restrict__ A, const __bf16* __restrict__ W,
                            const float* __restrict__ bias, float* __restrict__ C,
                            int M, int N, int K, int relu) {
  const int lane = threadIdx.x & 31;
  const int wv = threadIdx.x >> 5;
  const int hf = lane >> 4;
  const int l16 = lane & 15;
  const int Mw = blockIdx.x * 128 + (wv >> 1) * 32;
  const int Nw = blockIdx.y * 128 + (wv & 1) * 64;
  const float* Ar0 = A + (size_t)(Mw + l16) * K;
  const float* Ar1 = Ar0 + (size_t)16 * K;

  const __bf16* Wr[4];
  int ncol[4];
  bool nv[4];
#pragma unroll
  for (int j = 0; j < 4; j++) {
    int n = Nw + j * 16 + l16;
    nv[j] = n < N;
    ncol[j] = n;
    int nc = nv[j] ? n : (N - 1);  // clamp: garbage only lands in unstored columns
    Wr[j] = W + (size_t)nc * K + hf * 16;
  }

  v8f acc[2][4];
#pragma unroll
  for (int p = 0; p < 2; p++)
#pragma unroll
    for (int j = 0; j < 4; j++) acc[p][j] = (v8f){};

  for (int kb = 0; kb < K; kb += 32) {
    v16bf a0 = afrag_from_f32(Ar0 + kb + hf * 16);
    v16bf a1 = afrag_from_f32(Ar1 + kb + hf * 16);
    // B fragments: per lane 16 contiguous bf16 along K of its column
    v16bf b0 = *(const v16bf*)(Wr[0] + kb);
    v16bf b1 = *(const v16bf*)(Wr[1] + kb);
    v16bf b2 = *(const v16bf*)(Wr[2] + kb);
    v16bf b3 = *(const v16bf*)(Wr[3] + kb);
    acc[0][0] = wmma_bf16(a0, b0, acc[0][0]);
    acc[1][0] = wmma_bf16(a1, b0, acc[1][0]);
    acc[0][1] = wmma_bf16(a0, b1, acc[0][1]);
    acc[1][1] = wmma_bf16(a1, b1, acc[1][1]);
    acc[0][2] = wmma_bf16(a0, b2, acc[0][2]);
    acc[1][2] = wmma_bf16(a1, b2, acc[1][2]);
    acc[0][3] = wmma_bf16(a0, b3, acc[0][3]);
    acc[1][3] = wmma_bf16(a1, b3, acc[1][3]);
  }

#pragma unroll
  for (int j = 0; j < 4; j++) {
    if (!nv[j]) continue;
    float bv = bias ? bias[ncol[j]] : 0.0f;
#pragma unroll
    for (int p = 0; p < 2; p++) {
#pragma unroll
      for (int r = 0; r < 8; r++) {
        int row = Mw + p * 16 + r + hf * 8;
        float v = acc[p][j][r] + bv;
        if (relu) v = fmaxf(v, 0.0f);
        C[(size_t)row * N + ncol[j]] = v;
      }
    }
  }
}

// ---------------- repack qkv [S,B,3D] f32 -> Qh[h][s][64], Kh[h][s][64], Vt[h][64][s] bf16 --------
__global__ void repack_qkv_kernel(const float* __restrict__ qkv, __bf16* __restrict__ Qh,
                                  __bf16* __restrict__ Kh, __bf16* __restrict__ Vt, int S) {
  int idx = blockIdx.x * 256 + threadIdx.x;
  int d = idx & 63;
  int s = (idx >> 6) % S;
  int h = idx / (S * 64);
  if (h >= kH) return;
  int bcol = h >> 3, nh = h & 7;
  const float* p = qkv + ((size_t)s * kB + bcol) * (3 * kD) + nh * 64 + d;
  Qh[((size_t)h * S + s) * 64 + d] = (__bf16)p[0];
  Kh[((size_t)h * S + s) * 64 + d] = (__bf16)p[kD];
  Vt[((size_t)h * 64 + d) * S + s] = (__bf16)p[2 * kD];
}

// ---------------- repack x [S,B,D] f32 -> [h][s][64] or transposed [h][64][s] bf16 ----------------
__global__ void repack_x_kernel(const float* __restrict__ x, __bf16* __restrict__ dst,
                                int S, int transpose) {
  int idx = blockIdx.x * 256 + threadIdx.x;
  int d = idx & 63;
  int s = (idx >> 6) % S;
  int h = idx / (S * 64);
  if (h >= kH) return;
  int bcol = h >> 3, nh = h & 7;
  float v = x[((size_t)s * kB + bcol) * kD + nh * 64 + d];
  if (transpose)
    dst[((size_t)h * 64 + d) * S + s] = (__bf16)v;
  else
    dst[((size_t)h * S + s) * 64 + d] = (__bf16)v;
}

// ---------------- residual add + layernorm, one row per block ----------------
__global__ void add_ln_kernel(const float* __restrict__ x, const float* __restrict__ x2,
                              const float* __restrict__ w, const float* __restrict__ b,
                              float* __restrict__ y) {
  const int row = blockIdx.x;
  const int t = threadIdx.x;
  const size_t base = (size_t)row * kD;
  float v0 = x[base + t] + x2[base + t];
  float v1 = x[base + t + 256] + x2[base + t + 256];
  float s = v0 + v1, sq = v0 * v0 + v1 * v1;
#pragma unroll
  for (int m = 16; m >= 1; m >>= 1) {
    s += __shfl_xor(s, m, 32);
    sq += __shfl_xor(sq, m, 32);
  }
  __shared__ float ss[8], sqs[8];
  if ((t & 31) == 0) { ss[t >> 5] = s; sqs[t >> 5] = sq; }
  __syncthreads();
  s = 0.0f; sq = 0.0f;
#pragma unroll
  for (int i = 0; i < 8; i++) { s += ss[i]; sq += sqs[i]; }
  float mu = s * (1.0f / kD);
  float var = sq * (1.0f / kD) - mu * mu;
  float rstd = rsqrtf(var + kEPS);
  y[base + t] = (v0 - mu) * rstd * w[t] + b[t];
  y[base + t + 256] = (v1 - mu) * rstd * w[t + 256] + b[t + 256];
}

// ---------------- token embedding + sinusoidal positional encoding ----------------
__global__ void embed_kernel(const int* __restrict__ tokens, const float* __restrict__ tok_emb,
                             float* __restrict__ tgt) {
  int idx = blockIdx.x * 256 + threadIdx.x;  // SQ*B*D
  int d = idx & (kD - 1);
  int sb = idx >> 9;
  int bcol = sb & (kB - 1);
  int s = sb >> 4;
  if (s >= kSQ) return;
  int tok = tokens[s * kB + bcol];
  int dh = (d < 256) ? d : (d - 256);
  float dep = (float)dh * (1.0f / 256.0f);
  float ang = (float)s * __powf(10000.0f, -dep);
  float pe = (d < 256) ? __sinf(ang) : __cosf(ang);
  tgt[idx] = tok_emb[(size_t)tok * kD + d] + pe;
}

// ---------------- flash-style WMMA attention ----------------
// One wave per (head, 16-query tile). Qh[h][Sq][64], Kh[h][Sk][64], Vt[h][64][Sk] bf16.
// out written as [s*B + b][nh*64 + d] f32.
__global__ void attn_kernel(const __bf16* __restrict__ Qh, const __bf16* __restrict__ Kh,
                            const __bf16* __restrict__ Vt, float* __restrict__ out,
                            int Sq, int Sk, int causal, float scale) {
  __shared__ __bf16 lds_p[8][16 * 32];
  const int lane = threadIdx.x & 31;
  const int wv = threadIdx.x >> 5;
  const int hf = lane >> 4;
  const int l16 = lane & 15;
  const int qtiles = Sq >> 4;
  const int wid = blockIdx.x * 8 + wv;
  const int h = wid / qtiles;
  const int qt = wid - h * qtiles;
  if (h >= kH) return;
  const int bcol = h >> 3, nh = h & 7;

  // Q fragments (dk = 64 -> two 16x32 A-frags)
  const __bf16* qrow = Qh + ((size_t)h * Sq + qt * 16 + l16) * kDK;
  v16bf aq0, aq1;
#pragma unroll
  for (int i = 0; i < 8; i++) {
    aq0[i] = qrow[hf * 16 + i];
    aq0[i + 8] = qrow[hf * 16 + 8 + i];
    aq1[i] = qrow[32 + hf * 16 + i];
    aq1[i + 8] = qrow[40 + hf * 16 + i];
  }

  v8f o0 = {}, o1 = {}, o2 = {}, o3 = {};
  float rmax[8], rsum[8];
#pragma unroll
  for (int r = 0; r < 8; r++) { rmax[r] = -1e30f; rsum[r] = 0.0f; }

  const int kend = causal ? (qt * 16 + 16) : Sk;
  for (int kb = 0; kb < kend; kb += 32) {
    // scores: S = Q @ K^T for 32 keys (two 16x16 C tiles)
    v8f c0 = {}, c1 = {};
    {
      const __bf16* k0 = Kh + ((size_t)h * Sk + kb + l16) * kDK + hf * 16;
      const __bf16* k1 = k0 + (size_t)16 * kDK;
      v16bf b00 = *(const v16bf*)(k0);
      v16bf b01 = *(const v16bf*)(k0 + 32);
      v16bf b10 = *(const v16bf*)(k1);
      v16bf b11 = *(const v16bf*)(k1 + 32);
      c0 = wmma_bf16(aq0, b00, c0);
      c0 = wmma_bf16(aq1, b01, c0);
      c1 = wmma_bf16(aq0, b10, c1);
      c1 = wmma_bf16(aq1, b11, c1);
    }
    // online softmax update per row (row = r + hf*8, cols across half-wave)
#pragma unroll
    for (int r = 0; r < 8; r++) {
      float s0 = c0[r] * scale;
      float s1 = c1[r] * scale;
      if (causal) {
        int q = qt * 16 + r + hf * 8;
        if (kb + l16 > q) s0 = -1e30f;
        if (kb + 16 + l16 > q) s1 = -1e30f;
      }
      float m = redmax16(fmaxf(s0, s1));
      float nm = fmaxf(rmax[r], m);
      float fac = __expf(rmax[r] - nm);
      rmax[r] = nm;
      float p0 = __expf(s0 - nm);
      float p1 = __expf(s1 - nm);
      float ps = redsum16(p0 + p1);
      rsum[r] = rsum[r] * fac + ps;
      o0[r] *= fac; o1[r] *= fac; o2[r] *= fac; o3[r] *= fac;
      int mrow = r + hf * 8;
      lds_p[wv][mrow * 32 + l16] = (__bf16)p0;
      lds_p[wv][mrow * 32 + 16 + l16] = (__bf16)p1;
    }
    // C-layout -> A-layout transpose of P via per-wave LDS (same-wave DS ops are in-order)
    asm volatile("s_wait_dscnt 0" ::: "memory");
    v16bf pa;
    {
      const __bf16* pr = &lds_p[wv][l16 * 32];
#pragma unroll
      for (int i = 0; i < 8; i++) {
        pa[i] = pr[hf * 16 + i];
        pa[i + 8] = pr[hf * 16 + 8 + i];
      }
    }
    // O += P @ V  (V pre-transposed: contiguous along keys)
    {
      const __bf16* vb = Vt + ((size_t)h * kDK + l16) * Sk + kb + hf * 16;
      v16bf bv0 = *(const v16bf*)(vb);
      v16bf bv1 = *(const v16bf*)(vb + (size_t)16 * Sk);
      v16bf bv2 = *(const v16bf*)(vb + (size_t)32 * Sk);
      v16bf bv3 = *(const v16bf*)(vb + (size_t)48 * Sk);
      o0 = wmma_bf16(pa, bv0, o0);
      o1 = wmma_bf16(pa, bv1, o1);
      o2 = wmma_bf16(pa, bv2, o2);
      o3 = wmma_bf16(pa, bv3, o3);
    }
  }

#pragma unroll
  for (int r = 0; r < 8; r++) {
    int srow = qt * 16 + r + hf * 8;
    float inv = 1.0f / rsum[r];
    size_t base = ((size_t)srow * kB + bcol) * kD + (size_t)nh * 64;
    out[base + l16] = o0[r] * inv;
    out[base + 16 + l16] = o1[r] * inv;
    out[base + 32 + l16] = o2[r] * inv;
    out[base + 48 + l16] = o3[r] * inv;
  }
}

// =======================================================================================
extern "C" void kernel_launch(void* const* d_in, const int* in_sizes, int n_in,
                              void* d_out, int out_size, void* d_ws, size_t ws_size,
                              hipStream_t stream) {
  (void)in_sizes; (void)n_in; (void)out_size; (void)ws_size;
  const float* src = (const float*)d_in[0];
  const int* tokens = (const int*)d_in[1];
  const float* tok_emb = (const float*)d_in[2];
  const float* enc_qkv_w = (const float*)d_in[3];
  const float* enc_qkv_b = (const float*)d_in[4];
  const float* enc_out_w = (const float*)d_in[5];
  const float* enc_out_b = (const float*)d_in[6];
  const float* enc_l1_w = (const float*)d_in[7];
  const float* enc_l1_b = (const float*)d_in[8];
  const float* enc_l2_w = (const float*)d_in[9];
  const float* enc_l2_b = (const float*)d_in[10];
  const float* enc_n1_w = (const float*)d_in[11];
  const float* enc_n1_b = (const float*)d_in[12];
  const float* enc_n2_w = (const float*)d_in[13];
  const float* enc_n2_b = (const float*)d_in[14];
  const float* dec_qkv_w = (const float*)d_in[15];
  const float* dec_qkv_b = (const float*)d_in[16];
  const float* dec_sout_w = (const float*)d_in[17];
  const float* dec_sout_b = (const float*)d_in[18];
  const float* dec_q_w = (const float*)d_in[19];
  const float* dec_q_b = (const float*)d_in[20];
  const float* dec_k_w = (const float*)d_in[21];
  const float* dec_k_b = (const float*)d_in[22];
  const float* dec_v_w = (const float*)d_in[23];
  const float* dec_v_b = (const float*)d_in[24];
  const float* dec_cout_w = (const float*)d_in[25];
  const float* dec_cout_b = (const float*)d_in[26];
  const float* dec_l1_w = (const float*)d_in[27];
  const float* dec_l1_b = (const float*)d_in[28];
  const float* dec_l2_w = (const float*)d_in[29];
  const float* dec_l2_b = (const float*)d_in[30];
  const float* dec_n1_w = (const float*)d_in[31];
  const float* dec_n1_b = (const float*)d_in[32];
  const float* dec_n2_w = (const float*)d_in[33];
  const float* dec_n2_b = (const float*)d_in[34];
  const float* dec_n3_w = (const float*)d_in[35];
  const float* dec_n3_b = (const float*)d_in[36];
  const float* out_w = (const float*)d_in[37];
  const float* out_b = (const float*)d_in[38];

  char* wsp = (char*)d_ws;
  size_t off = 0;
  auto alloc = [&](size_t bytes) -> void* {
    void* p = wsp + off;
    off += (bytes + 255) & ~(size_t)255;
    return p;
  };
  auto cvtw = [&](const float* s, size_t n) -> __bf16* {
    __bf16* dst = (__bf16*)alloc(n * 2);
    cvt_kernel<<<(int)((n + 255) / 256), 256, 0, stream>>>(s, dst, (int)n);
    return dst;
  };
  auto gemm = [&](const float* A, const __bf16* W, const float* bias, float* C,
                  int M, int N, int K, int relu) {
    dim3 grid(M / 128, (N + 127) / 128);
    gemm_kernel<<<grid, 256, 0, stream>>>(A, W, bias, C, M, N, K, relu);
  };
  auto addln = [&](const float* x_, const float* x2_, const float* w_, const float* b_,
                   float* y_, int rows) {
    add_ln_kernel<<<rows, 256, 0, stream>>>(x_, x2_, w_, b_, y_);
  };
  auto attn = [&](const __bf16* Qh_, const __bf16* Kh_, const __bf16* Vt_, float* o_,
                  int Sq, int Sk, int causal) {
    int waves = kH * (Sq / 16);
    attn_kernel<<<waves / 8, 256, 0, stream>>>(Qh_, Kh_, Vt_, o_, Sq, Sk, causal, 0.125f);
  };

  // ---- weights -> bf16 (once per call; deterministic) ----
  __bf16* w_eqkv = cvtw(enc_qkv_w, (size_t)kLE * 3 * kD * kD);
  __bf16* w_eout = cvtw(enc_out_w, (size_t)kLE * kD * kD);
  __bf16* w_el1 = cvtw(enc_l1_w, (size_t)kLE * kFF * kD);
  __bf16* w_el2 = cvtw(enc_l2_w, (size_t)kLE * kD * kFF);
  __bf16* w_dqkv = cvtw(dec_qkv_w, (size_t)kLD * 3 * kD * kD);
  __bf16* w_dsout = cvtw(dec_sout_w, (size_t)kLD * kD * kD);
  __bf16* w_dq = cvtw(dec_q_w, (size_t)kLD * kD * kD);
  __bf16* w_dk = cvtw(dec_k_w, (size_t)kLD * kD * kD);
  __bf16* w_dv = cvtw(dec_v_w, (size_t)kLD * kD * kD);
  __bf16* w_dcout = cvtw(dec_cout_w, (size_t)kLD * kD * kD);
  __bf16* w_dl1 = cvtw(dec_l1_w, (size_t)kLD * kFF * kD);
  __bf16* w_dl2 = cvtw(dec_l2_w, (size_t)kLD * kD * kFF);
  __bf16* w_o = cvtw(out_w, (size_t)kV * kD);

  // ---- activations ----
  const int Me = kSM * kB;  // 8192 encoder rows
  const int Md = kSQ * kB;  // 2048 decoder rows
  float* x = (float*)alloc((size_t)Me * kD * 4);
  float* x2 = (float*)alloc((size_t)Me * kD * 4);
  float* qkv = (float*)alloc((size_t)Me * 3 * kD * 4);
  float* ffh = (float*)alloc((size_t)Me * kFF * 4);
  float* attnb = (float*)alloc((size_t)Me * kD * 4);
  __bf16* Qh = (__bf16*)alloc((size_t)kH * kSM * 64 * 2);
  __bf16* Kh = (__bf16*)alloc((size_t)kH * kSM * 64 * 2);
  __bf16* Vt = (__bf16*)alloc((size_t)kH * 64 * kSM * 2);
  float* tgt = (float*)alloc((size_t)Md * kD * 4);
  float* tgt2 = (float*)alloc((size_t)Md * kD * 4);

  // ---- encoder ----
  const float* cur = src;
  for (int i = 0; i < kLE; i++) {
    gemm(cur, w_eqkv + (size_t)i * 3 * kD * kD, enc_qkv_b + (size_t)i * 3 * kD, qkv,
         Me, 3 * kD, kD, 0);
    repack_qkv_kernel<<<(kH * kSM * 64) / 256, 256, 0, stream>>>(qkv, Qh, Kh, Vt, kSM);
    attn(Qh, Kh, Vt, attnb, kSM, kSM, 0);
    gemm(attnb, w_eout + (size_t)i * kD * kD, enc_out_b + (size_t)i * kD, x2, Me, kD, kD, 0);
    addln(cur, x2, enc_n1_w + (size_t)i * kD, enc_n1_b + (size_t)i * kD, x, Me);
    gemm(x, w_el1 + (size_t)i * kFF * kD, enc_l1_b + (size_t)i * kFF, ffh, Me, kFF, kD, 1);
    gemm(ffh, w_el2 + (size_t)i * kD * kFF, enc_l2_b + (size_t)i * kD, x2, Me, kD, kFF, 0);
    addln(x, x2, enc_n2_w + (size_t)i * kD, enc_n2_b + (size_t)i * kD, x, Me);
    cur = x;
  }
  // x now holds encoder memory [SM,B,D]

  // ---- decoder ----
  embed_kernel<<<(kSQ * kB * kD) / 256, 256, 0, stream>>>(tokens, tok_emb, tgt);
  float* qx = qkv;                       // [Md,kD]
  float* kx = qkv + (size_t)Md * kD;     // [Me,kD]
  float* vx = kx + (size_t)Me * kD;      // [Me,kD]
  for (int i = 0; i < kLD; i++) {
    // causal self-attention
    gemm(tgt, w_dqkv + (size_t)i * 3 * kD * kD, dec_qkv_b + (size_t)i * 3 * kD, qkv,
         Md, 3 * kD, kD, 0);
    repack_qkv_kernel<<<(kH * kSQ * 64) / 256, 256, 0, stream>>>(qkv, Qh, Kh, Vt, kSQ);
    attn(Qh, Kh, Vt, attnb, kSQ, kSQ, 1);
    gemm(attnb, w_dsout + (size_t)i * kD * kD, dec_sout_b + (size_t)i * kD, tgt2, Md, kD, kD, 0);
    addln(tgt, tgt2, dec_n1_w + (size_t)i * kD, dec_n1_b + (size_t)i * kD, tgt, Md);
    // cross-attention against encoder memory
    gemm(tgt, w_dq + (size_t)i * kD * kD, dec_q_b + (size_t)i * kD, qx, Md, kD, kD, 0);
    gemm(x, w_dk + (size_t)i * kD * kD, dec_k_b + (size_t)i * kD, kx, Me, kD, kD, 0);
    gemm(x, w_dv + (size_t)i * kD * kD, dec_v_b + (size_t)i * kD, vx, Me, kD, kD, 0);
    repack_x_kernel<<<(kH * kSQ * 64) / 256, 256, 0, stream>>>(qx, Qh, kSQ, 0);
    repack_x_kernel<<<(kH * kSM * 64) / 256, 256, 0, stream>>>(kx, Kh, kSM, 0);
    repack_x_kernel<<<(kH * kSM * 64) / 256, 256, 0, stream>>>(vx, Vt, kSM, 1);
    attn(Qh, Kh, Vt, attnb, kSQ, kSM, 0);
    gemm(attnb, w_dcout + (size_t)i * kD * kD, dec_cout_b + (size_t)i * kD, tgt2, Md, kD, kD, 0);
    addln(tgt, tgt2, dec_n2_w + (size_t)i * kD, dec_n2_b + (size_t)i * kD, tgt, Md);
    // FFN
    gemm(tgt, w_dl1 + (size_t)i * kFF * kD, dec_l1_b + (size_t)i * kFF, ffh, Md, kFF, kD, 1);
    gemm(ffh, w_dl2 + (size_t)i * kD * kFF, dec_l2_b + (size_t)i * kD, tgt2, Md, kD, kFF, 0);
    addln(tgt, tgt2, dec_n3_w + (size_t)i * kD, dec_n3_b + (size_t)i * kD, tgt, Md);
  }

  // ---- logits [SQ*B, V] ----
  gemm(tgt, w_o, out_b, (float*)d_out, Md, kV, kD, 0);
}
